// PointConvSceneFlowPWC_17351667876610
// MI455X (gfx1250) — matrix operand order repaired and checked by reading, not compiled, not run on hardware
//
#include <hip/hip_runtime.h>
#include <cstddef>
#include <cstdint>

// ---------------------------------------------------------------------------
// CDNA5 (gfx1250) implementation of PointPWC-Net scene-flow forward pass.
// All dense linear algebra goes through v_wmma_f32_16x16x32_bf16 (wave32).
// Irregular point ops (knn/fps/gather/interp) are plain VALU kernels.
// ---------------------------------------------------------------------------

#define DEVI __device__ __forceinline__

typedef __attribute__((ext_vector_type(16))) __bf16 v16bf;
typedef __attribute__((ext_vector_type(8)))  float  v8f;
// 4-float vector with relaxed (4-byte) alignment: rows may have odd stride.
typedef float f4a __attribute__((ext_vector_type(4), aligned(4)));

DEVI __bf16 f2bf(float f) {
  // round-to-nearest-even fp32 -> bf16, via integer path
  unsigned u = __builtin_bit_cast(unsigned, f);
  unsigned r = (u + 0x7FFFu + ((u >> 16) & 1u)) >> 16;
  unsigned short s = (unsigned short)r;
  return __builtin_bit_cast(__bf16, s);
}

DEVI float act_apply(float v, int act) {
  if (act == 1) return (v > 0.0f) ? v : 0.1f * v;        // LeakyReLU(0.1)
  if (act == 2) return fmaxf(v, 0.0f);                   // ReLU
  if (act == 3) return fminf(fmaxf(v, -200.0f), 200.0f); // clip +-200
  return v;
}

DEVI void cvt8(const f4a x, const f4a y, v16bf& dst, int base) {
#pragma unroll
  for (int i = 0; i < 4; ++i) dst[base + i] = f2bf(x[i]);
#pragma unroll
  for (int i = 0; i < 4; ++i) dst[base + 4 + i] = f2bf(y[i]);
}

// ---------------------------------------------------------------------------
// WMMA GEMM: out[rows, cout] = act(in[rows, cin] * W[cout, cin]^T + b)
// One wave computes a 16x32 output tile (one A fragment, two B fragments,
// two accumulators), K-step = 32 (bf16). Branch-free main loop: OOB rows/cols
// are clamped to row/col 0 (loads valid, results discarded at store time).
// Fragment layouts used here (contiguous k; A and B share it):
//   A 16x32: lane half h, row m=lane&15; af[i] = A[m][h*16+i]
//            (one contiguous 16-float run per lane)
//   B 32x16: lane half h, col n=lane&15; bf[i] = B[h*16+i][n]
//            (one contiguous 16-float run of W[n][:])
//   C/D 32b: row = vgpr + 8*(lane>>4), col = lane&15
// ---------------------------------------------------------------------------
__global__ void wmma_linear_kernel(const float* __restrict__ in,
                                   const float* __restrict__ W,
                                   const float* __restrict__ b,
                                   float* __restrict__ out,
                                   int rows, int cin, int cout, int act) {
  int tm = blockIdx.x;
  int tn = blockIdx.y;
  int lane = threadIdx.x & 31;
  int half = lane >> 4;
  int lm = lane & 15;

  int row = tm * 16 + lm;
  int col0 = tn * 32 + lm;
  int col1 = col0 + 16;

  const float* arow  = in + (size_t)(row  < rows ? row  : 0) * cin;
  const float* wrow0 = W  + (size_t)(col0 < cout ? col0 : 0) * cin;
  const float* wrow1 = W  + (size_t)(col1 < cout ? col1 : 0) * cin;

  v8f acc0 = {};
  v8f acc1 = {};
  int kmain = cin & ~31;

  for (int k0 = 0; k0 < kmain; k0 += 32) {
    if (k0 + 64 <= kmain) __builtin_prefetch(arow + k0 + 64, 0, 3);
    const float* ap = arow + k0 + half * 16;
    f4a a0 = *(const f4a*)(ap);
    f4a a1 = *(const f4a*)(ap + 4);
    f4a a2 = *(const f4a*)(ap + 8);
    f4a a3 = *(const f4a*)(ap + 12);
    const float* bp0 = wrow0 + k0 + half * 16;
    f4a b00 = *(const f4a*)(bp0);
    f4a b01 = *(const f4a*)(bp0 + 4);
    f4a b02 = *(const f4a*)(bp0 + 8);
    f4a b03 = *(const f4a*)(bp0 + 12);
    const float* bp1 = wrow1 + k0 + half * 16;
    f4a b10 = *(const f4a*)(bp1);
    f4a b11 = *(const f4a*)(bp1 + 4);
    f4a b12 = *(const f4a*)(bp1 + 8);
    f4a b13 = *(const f4a*)(bp1 + 12);

    v16bf af, bf0, bf1;
    cvt8(a0, a1, af, 0);
    cvt8(a2, a3, af, 8);
    cvt8(b00, b01, bf0, 0);
    cvt8(b02, b03, bf0, 8);
    cvt8(b10, b11, bf1, 0);
    cvt8(b12, b13, bf1, 8);

    acc0 = __builtin_amdgcn_wmma_f32_16x16x32_bf16(false, af, false, bf0,
                                                   (short)0, acc0, false, false);
    acc1 = __builtin_amdgcn_wmma_f32_16x16x32_bf16(false, af, false, bf1,
                                                   (short)0, acc1, false, false);
  }

  if (kmain < cin) {
    // Single remainder step, branch-free: clamp index + select 0.
    v16bf af, bf0, bf1;
#pragma unroll
    for (int i = 0; i < 16; ++i) {
      int k = half * 16 + i;
      int kk = kmain + k;
      int kc = kk < cin ? kk : 0;
      float v = arow[kc];
      af[i] = f2bf(kk < cin ? v : 0.0f);
    }
#pragma unroll
    for (int i = 0; i < 16; ++i) {
      int kk = kmain + half * 16 + i;
      int kc = kk < cin ? kk : 0;
      float v0 = wrow0[kc];
      float v1 = wrow1[kc];
      bf0[i] = f2bf(kk < cin ? v0 : 0.0f);
      bf1[i] = f2bf(kk < cin ? v1 : 0.0f);
    }
    acc0 = __builtin_amdgcn_wmma_f32_16x16x32_bf16(false, af, false, bf0,
                                                   (short)0, acc0, false, false);
    acc1 = __builtin_amdgcn_wmma_f32_16x16x32_bf16(false, af, false, bf1,
                                                   (short)0, acc1, false, false);
  }

  float bias0 = b ? b[col0 < cout ? col0 : 0] : 0.0f;
  float bias1 = b ? b[col1 < cout ? col1 : 0] : 0.0f;
#pragma unroll
  for (int r = 0; r < 8; ++r) {
    int m = tm * 16 + r + half * 8;
    if (m < rows) {
      if (col0 < cout) out[(size_t)m * cout + col0] = act_apply(acc0[r] + bias0, act);
      if (col1 < cout) out[(size_t)m * cout + col1] = act_apply(acc1[r] + bias1, act);
    }
  }
}

// ---------------------------------------------------------------------------
// Brute-force KNN: per-query register insertion sort of K best.
// ---------------------------------------------------------------------------
template <int K>
__global__ void knn_kernel(const float* __restrict__ ref, int M,
                           const float* __restrict__ query, int Nq,
                           int* __restrict__ idx) {
  int n = blockIdx.x * blockDim.x + threadIdx.x;
  if (n >= Nq) return;
  float qx = query[n * 3], qy = query[n * 3 + 1], qz = query[n * 3 + 2];
  float bd[K];
  int bi[K];
#pragma unroll
  for (int i = 0; i < K; ++i) { bd[i] = 3.4e38f; bi[i] = 0; }
  for (int m = 0; m < M; ++m) {
    float dx = ref[m * 3] - qx;
    float dy = ref[m * 3 + 1] - qy;
    float dz = ref[m * 3 + 2] - qz;
    float d = dx * dx + dy * dy + dz * dz;
    if (d < bd[K - 1]) {
      int p = K - 1;
      while (p > 0 && bd[p - 1] > d) {
        bd[p] = bd[p - 1]; bi[p] = bi[p - 1]; --p;
      }
      bd[p] = d; bi[p] = m;
    }
  }
  for (int i = 0; i < K; ++i) idx[(size_t)n * K + i] = bi[i];
}

// ---------------------------------------------------------------------------
// Farthest point sampling (single block, iterative argmax reduction).
// ---------------------------------------------------------------------------
__global__ void fps_kernel(const float* __restrict__ xyz, int N, int npoint,
                           int* __restrict__ out_idx, float* __restrict__ dist) {
  __shared__ float sval[256];
  __shared__ int   sidx[256];
  __shared__ int   s_far;
  int t = threadIdx.x;
  for (int i = t; i < N; i += blockDim.x) dist[i] = 1e10f;
  if (t == 0) s_far = 0;
  __syncthreads();
  for (int it = 0; it < npoint; ++it) {
    int far = s_far;
    if (t == 0) out_idx[it] = far;
    float cx = xyz[far * 3], cy = xyz[far * 3 + 1], cz = xyz[far * 3 + 2];
    float best = -1.0f; int bidx = 0;
    for (int i = t; i < N; i += blockDim.x) {
      float dx = xyz[i * 3] - cx, dy = xyz[i * 3 + 1] - cy, dz = xyz[i * 3 + 2] - cz;
      float d = dx * dx + dy * dy + dz * dz;
      float dd = fminf(dist[i], d);
      dist[i] = dd;
      if (dd > best) { best = dd; bidx = i; }
    }
    sval[t] = best; sidx[t] = bidx;
    __syncthreads();
    for (int s = 128; s > 0; s >>= 1) {
      if (t < s && sval[t + s] > sval[t]) { sval[t] = sval[t + s]; sidx[t] = sidx[t + s]; }
      __syncthreads();
    }
    if (t == 0) s_far = sidx[0];
    __syncthreads();
  }
}

// ---------------------------------------------------------------------------
// Small pointwise / gather kernels.
// ---------------------------------------------------------------------------
__global__ void gather_rows_kernel(const float* __restrict__ src, int C,
                                   const int* __restrict__ idx, int S,
                                   float* __restrict__ out) {
  int i = blockIdx.x * blockDim.x + threadIdx.x;
  if (i >= S * C) return;
  int s = i / C, c = i % C;
  out[i] = src[(size_t)idx[s] * C + c];
}

__global__ void group_concat_kernel(const float* __restrict__ xyz,
                                    const float* __restrict__ feats, int Cf,
                                    const int* __restrict__ idx,
                                    const float* __restrict__ qxyz,
                                    int S, int K,
                                    float* __restrict__ gx,
                                    float* __restrict__ gf) {
  int i = blockIdx.x * blockDim.x + threadIdx.x;
  if (i >= S * K) return;
  int s = i / K;
  int m = idx[i];
  float dx = xyz[m * 3] - qxyz[s * 3];
  float dy = xyz[m * 3 + 1] - qxyz[s * 3 + 1];
  float dz = xyz[m * 3 + 2] - qxyz[s * 3 + 2];
  gx[(size_t)i * 3] = dx; gx[(size_t)i * 3 + 1] = dy; gx[(size_t)i * 3 + 2] = dz;
  float* o = gf + (size_t)i * (3 + Cf);
  o[0] = dx; o[1] = dy; o[2] = dz;
  const float* fr = feats + (size_t)m * Cf;
  for (int c = 0; c < Cf; ++c) o[3 + c] = fr[c];
}

__global__ void dir_kernel(const float* __restrict__ ref,
                           const float* __restrict__ q,
                           const int* __restrict__ idx,
                           int n0, int nc, int K, float* __restrict__ dir) {
  int i = blockIdx.x * blockDim.x + threadIdx.x;
  if (i >= nc * K) return;
  int n = n0 + i / K;
  int m = idx[(size_t)n * K + (i % K)];
  dir[(size_t)i * 3]     = ref[m * 3]     - q[n * 3];
  dir[(size_t)i * 3 + 1] = ref[m * 3 + 1] - q[n * 3 + 1];
  dir[(size_t)i * 3 + 2] = ref[m * 3 + 2] - q[n * 3 + 2];
}

// WeightNet MLP 3 -> 8 -> 8 -> Cout, ReLU after every layer.
__global__ void weightnet_kernel(const float* __restrict__ xyzrel, int Npts,
                                 const float* __restrict__ w0, const float* __restrict__ b0,
                                 const float* __restrict__ w1, const float* __restrict__ b1,
                                 const float* __restrict__ w2, const float* __restrict__ b2,
                                 int Cout, float* __restrict__ out) {
  int n = blockIdx.x * blockDim.x + threadIdx.x;
  if (n >= Npts) return;
  float x0 = xyzrel[n * 3], x1 = xyzrel[n * 3 + 1], x2 = xyzrel[n * 3 + 2];
  float h[8], g[8];
#pragma unroll
  for (int o = 0; o < 8; ++o) {
    float a = b0[o] + w0[o * 3] * x0 + w0[o * 3 + 1] * x1 + w0[o * 3 + 2] * x2;
    h[o] = fmaxf(a, 0.0f);
  }
#pragma unroll
  for (int o = 0; o < 8; ++o) {
    float a = b1[o];
#pragma unroll
    for (int j = 0; j < 8; ++j) a += w1[o * 8 + j] * h[j];
    g[o] = fmaxf(a, 0.0f);
  }
  float* op = out + (size_t)n * Cout;
  for (int o = 0; o < Cout; ++o) {
    float a = b2[o];
#pragma unroll
    for (int j = 0; j < 8; ++j) a += w2[o * 8 + j] * g[j];
    op[o] = fmaxf(a, 0.0f);
  }
}

// agg[s, c*16 + w] = sum_k gf[s,k,c] * wn[s,k,w]   (K <= 16)
__global__ void pconv_agg_kernel(const float* __restrict__ gf,
                                 const float* __restrict__ wn,
                                 int S, int K, int C, float* __restrict__ agg) {
  __shared__ float sw[16 * 16];
  int s = blockIdx.x;
  int t = threadIdx.x;
  for (int i = t; i < K * 16; i += blockDim.x) sw[i] = wn[(size_t)s * K * 16 + i];
  __syncthreads();
  const float* g = gf + (size_t)s * K * C;
  for (int i = t; i < C * 16; i += blockDim.x) {
    int c = i >> 4, w = i & 15;
    float a = 0.0f;
    for (int k = 0; k < K; ++k) a += g[(size_t)k * C + c] * sw[k * 16 + w];
    agg[(size_t)s * C * 16 + i] = a;
  }
}

// out[n0+n, c] = sum_k w[n,k,c] * x[n,k,c]   (w,x local to chunk)
__global__ void mulsum_k_kernel(const float* __restrict__ w,
                                const float* __restrict__ x,
                                int n0, int nc, int K, int C,
                                float* __restrict__ out) {
  int i = blockIdx.x * blockDim.x + threadIdx.x;
  if (i >= nc * C) return;
  int n = i / C, c = i % C;
  float a = 0.0f;
  for (int k = 0; k < K; ++k) {
    size_t o = ((size_t)n * K + k) * C + c;
    a += w[o] * x[o];
  }
  out[(size_t)(n0 + n) * C + c] = a;
}

// out[n0+n, c] = sum_k w[n,k,c] * feat[idx[n0+n,k], c]
__global__ void gws_kernel(const float* __restrict__ w,
                           const float* __restrict__ feat,
                           const int* __restrict__ idx,
                           int n0, int nc, int K, int C,
                           float* __restrict__ out) {
  int i = blockIdx.x * blockDim.x + threadIdx.x;
  if (i >= nc * C) return;
  int n = i / C, c = i % C;
  float a = 0.0f;
  for (int k = 0; k < K; ++k) {
    int m = idx[(size_t)(n0 + n) * K + k];
    a += w[((size_t)n * K + k) * C + c] * feat[(size_t)m * C + c];
  }
  out[(size_t)(n0 + n) * C + c] = a;
}

// 3-NN inverse-distance interpolation; out = base + sign * interp
__global__ void interp3_kernel(const float* __restrict__ dxyz,
                               const float* __restrict__ sxyz,
                               const int* __restrict__ idx,
                               const float* __restrict__ sfeat,
                               int C, int N,
                               const float* __restrict__ base, float sign,
                               float* __restrict__ out) {
  int i = blockIdx.x * blockDim.x + threadIdx.x;
  if (i >= N * C) return;
  int n = i / C, c = i % C;
  float qx = dxyz[n * 3], qy = dxyz[n * 3 + 1], qz = dxyz[n * 3 + 2];
  float w[3]; int id[3]; float wsum = 0.0f;
#pragma unroll
  for (int j = 0; j < 3; ++j) {
    int m = idx[n * 3 + j];
    id[j] = m;
    float dx = sxyz[m * 3] - qx, dy = sxyz[m * 3 + 1] - qy, dz = sxyz[m * 3 + 2] - qz;
    float d = sqrtf(dx * dx + dy * dy + dz * dz);
    d = fmaxf(d, 1e-10f);
    w[j] = 1.0f / d;
    wsum += w[j];
  }
  float a = 0.0f;
#pragma unroll
  for (int j = 0; j < 3; ++j) a += (w[j] / wsum) * sfeat[(size_t)id[j] * C + c];
  out[i] = (base ? base[i] : 0.0f) + sign * a;
}

__global__ void concat2_kernel(const float* __restrict__ a, int Ca,
                               const float* __restrict__ b, int Cb,
                               int N, float* __restrict__ out) {
  int C = Ca + Cb;
  int i = blockIdx.x * blockDim.x + threadIdx.x;
  if (i >= N * C) return;
  int n = i / C, ch = i % C;
  out[i] = (ch < Ca) ? a[(size_t)n * Ca + ch] : b[(size_t)n * Cb + (ch - Ca)];
}

__global__ void concat3_kernel(const float* __restrict__ a, int Ca,
                               const float* __restrict__ b, int Cb,
                               const float* __restrict__ c, int Cc,
                               int N, float* __restrict__ out) {
  int C = Ca + Cb + Cc;
  int i = blockIdx.x * blockDim.x + threadIdx.x;
  if (i >= N * C) return;
  int n = i / C, ch = i % C;
  float v;
  if (ch < Ca) v = a[(size_t)n * Ca + ch];
  else if (ch < Ca + Cb) v = b[(size_t)n * Cb + (ch - Ca)];
  else v = c[(size_t)n * Cc + (ch - Ca - Cb)];
  out[i] = v;
}

// Cost-volume input build: x = [f1[n], f2[idx], dir], dir also stored.
__global__ void cv_build_kernel(const float* __restrict__ xyz1,
                                const float* __restrict__ xyz2,
                                const float* __restrict__ f1,
                                const float* __restrict__ f2, int C,
                                const int* __restrict__ idx,
                                int n0, int nc, int K,
                                float* __restrict__ x, float* __restrict__ dir) {
  int i = blockIdx.x * blockDim.x + threadIdx.x;
  if (i >= nc * K) return;
  int n = n0 + i / K;
  int m = idx[(size_t)n * K + (i % K)];
  float dx = xyz2[m * 3] - xyz1[n * 3];
  float dy = xyz2[m * 3 + 1] - xyz1[n * 3 + 1];
  float dz = xyz2[m * 3 + 2] - xyz1[n * 3 + 2];
  int CT = 2 * C + 3;
  float* o = x + (size_t)i * CT;
  const float* a = f1 + (size_t)n * C;
  const float* b = f2 + (size_t)m * C;
  for (int c = 0; c < C; ++c) o[c] = a[c];
  for (int c = 0; c < C; ++c) o[C + c] = b[c];
  o[2 * C] = dx; o[2 * C + 1] = dy; o[2 * C + 2] = dz;
  dir[(size_t)i * 3] = dx; dir[(size_t)i * 3 + 1] = dy; dir[(size_t)i * 3 + 2] = dz;
}

__global__ void add_kernel(const float* __restrict__ a, const float* __restrict__ b,
                           int n, float* __restrict__ out) {
  int i = blockIdx.x * blockDim.x + threadIdx.x;
  if (i < n) out[i] = a[i] + b[i];
}

// ---------------------------------------------------------------------------
// Host-side orchestration
// ---------------------------------------------------------------------------
namespace {

struct Lin { const float* b; const float* w; };
struct WN { Lin l[3]; };
struct PConv { const float* lin_b; const float* lin_w; WN wn; };
struct PCF { Lin mlp[2]; WN wn1, wn2; };
struct SFEp { Lin fc; Lin mlp[2]; PConv pc[2]; };

struct WS {
  char* base; size_t cap; size_t off;
  float* f(size_t n) {
    size_t bytes = ((n * sizeof(float)) + 255) & ~(size_t)255;
    if (off + bytes > cap) off = 0;  // last-resort wrap (avoid OOB writes)
    char* p = base + off;
    off += bytes;
    return (float*)p;
  }
  int* i(size_t n) { return (int*)f(n); }
};

inline int cdiv(int a, int b) { return (a + b - 1) / b; }
inline int hmin(int a, int b) { return a < b ? a : b; }

void linear(hipStream_t st, const float* in, int rows, int cin, const Lin& L,
            int cout, float* out, int act) {
  dim3 g(cdiv(rows, 16), cdiv(cout, 32));
  wmma_linear_kernel<<<g, 32, 0, st>>>(in, L.w, L.b, out, rows, cin, cout, act);
}

void knn(hipStream_t st, int k, const float* ref, int M, const float* q, int Nq,
         int* idx) {
  int thr = 128, blk = cdiv(Nq, thr);
  switch (k) {
    case 3:  knn_kernel<3><<<blk, thr, 0, st>>>(ref, M, q, Nq, idx); break;
    case 9:  knn_kernel<9><<<blk, thr, 0, st>>>(ref, M, q, Nq, idx); break;
    case 16: knn_kernel<16><<<blk, thr, 0, st>>>(ref, M, q, Nq, idx); break;
    case 32: knn_kernel<32><<<blk, thr, 0, st>>>(ref, M, q, Nq, idx); break;
    default: break;
  }
}

void weightnet(hipStream_t st, const WN& w, const float* xyzrel, int npts,
               int cout, float* out) {
  weightnet_kernel<<<cdiv(npts, 128), 128, 0, st>>>(
      xyzrel, npts, w.l[0].w, w.l[0].b, w.l[1].w, w.l[1].b, w.l[2].w, w.l[2].b,
      cout, out);
}

// PointConv core, chunked over query points.
void pconv_core(hipStream_t st, WS& ws, const PConv& p, const float* xyz,
                const float* feats, int Cf, const int* idx, const float* qxyz,
                int S, int K, int cout, float* out) {
  int C = 3 + Cf;
  int chunk = 2048;
  for (int s0 = 0; s0 < S; s0 += chunk) {
    int sc = hmin(chunk, S - s0);
    size_t mark = ws.off;
    float* gx = ws.f((size_t)sc * K * 3);
    float* gf = ws.f((size_t)sc * K * C);
    group_concat_kernel<<<cdiv(sc * K, 128), 128, 0, st>>>(
        xyz, feats, Cf, idx + (size_t)s0 * K, qxyz + (size_t)s0 * 3, sc, K, gx, gf);
    float* wb = ws.f((size_t)sc * K * 16);
    weightnet(st, p.wn, gx, sc * K, 16, wb);
    float* agg = ws.f((size_t)sc * C * 16);
    pconv_agg_kernel<<<sc, 256, 0, st>>>(gf, wb, sc, K, C, agg);
    Lin L{p.lin_b, p.lin_w};
    linear(st, agg, sc, C * 16, L, cout, out + (size_t)s0 * cout, 1);
    ws.off = mark;
  }
}

void pointconv_d(hipStream_t st, WS& ws, const PConv& p, int npoint, int K,
                 const float* xyz, int N, const float* feats, int Cf, int cout,
                 float* new_xyz, float* out) {
  size_t mark = ws.off;
  int* fpsidx = ws.i(npoint);
  float* dist = ws.f(N);
  fps_kernel<<<1, 256, 0, st>>>(xyz, N, npoint, fpsidx, dist);
  gather_rows_kernel<<<cdiv(npoint * 3, 128), 128, 0, st>>>(xyz, 3, fpsidx, npoint, new_xyz);
  int* idx = ws.i((size_t)npoint * K);
  knn(st, K, xyz, N, new_xyz, npoint, idx);
  pconv_core(st, ws, p, xyz, feats, Cf, idx, new_xyz, npoint, K, cout, out);
  ws.off = mark;
}

void upsample(hipStream_t st, WS& ws, const float* dxyz, int Nd,
              const float* sxyz, int Ns, const float* sfeat, int C, float* out) {
  size_t mark = ws.off;
  int* idx = ws.i((size_t)Nd * 3);
  knn(st, 3, sxyz, Ns, dxyz, Nd, idx);
  interp3_kernel<<<cdiv(Nd * C, 128), 128, 0, st>>>(dxyz, sxyz, idx, sfeat, C, Nd,
                                                    nullptr, 1.0f, out);
  ws.off = mark;
}

void warp(hipStream_t st, WS& ws, const float* xyz1, int N1, const float* xyz2,
          int N2, const float* flow1, float* out) {
  size_t mark = ws.off;
  float* x12 = ws.f((size_t)N1 * 3);
  add_kernel<<<cdiv(N1 * 3, 128), 128, 0, st>>>(xyz1, flow1, N1 * 3, x12);
  int* idx = ws.i((size_t)N2 * 3);
  knn(st, 3, x12, N1, xyz2, N2, idx);
  interp3_kernel<<<cdiv(N2 * 3, 128), 128, 0, st>>>(xyz2, x12, idx, flow1, 3, N2,
                                                    xyz2, -1.0f, out);
  ws.off = mark;
}

void pcflow(hipStream_t st, WS& ws, const PCF& p, int K, const float* xyz1,
            int N1, const float* xyz2, int N2, const float* f1, const float* f2,
            int C, int m0, int m1, float* out) {
  size_t mark = ws.off;
  int* idx = ws.i((size_t)N1 * K);
  knn(st, K, xyz2, N2, xyz1, N1, idx);
  float* p2p = ws.f((size_t)N1 * m1);
  int CT = 2 * C + 3;
  int chunk = hmin(N1, (16384 / K > 0) ? (16384 / K) : 1);
  for (int n0 = 0; n0 < N1; n0 += chunk) {
    int nc = hmin(chunk, N1 - n0);
    size_t m2 = ws.off;
    int rows = nc * K;
    float* x = ws.f((size_t)rows * CT);
    float* dir = ws.f((size_t)rows * 3);
    cv_build_kernel<<<cdiv(rows, 128), 128, 0, st>>>(xyz1, xyz2, f1, f2, C, idx,
                                                     n0, nc, K, x, dir);
    float* h0 = ws.f((size_t)rows * m0);
    linear(st, x, rows, CT, p.mlp[0], m0, h0, 1);
    float* h1 = ws.f((size_t)rows * m1);
    linear(st, h0, rows, m0, p.mlp[1], m1, h1, 1);
    float* w1 = ws.f((size_t)rows * m1);
    weightnet(st, p.wn1, dir, rows, m1, w1);
    mulsum_k_kernel<<<cdiv(nc * m1, 128), 128, 0, st>>>(w1, h1, n0, nc, K, m1, p2p);
    ws.off = m2;
  }
  int* idx2 = ws.i((size_t)N1 * K);
  knn(st, K, xyz1, N1, xyz1, N1, idx2);
  for (int n0 = 0; n0 < N1; n0 += chunk) {
    int nc = hmin(chunk, N1 - n0);
    size_t m2 = ws.off;
    int rows = nc * K;
    float* dir2 = ws.f((size_t)rows * 3);
    dir_kernel<<<cdiv(rows, 128), 128, 0, st>>>(xyz1, xyz1, idx2, n0, nc, K, dir2);
    float* w2 = ws.f((size_t)rows * m1);
    weightnet(st, p.wn2, dir2, rows, m1, w2);
    gws_kernel<<<cdiv(nc * m1, 128), 128, 0, st>>>(w2, p2p, idx2, n0, nc, K, m1, out);
    ws.off = m2;
  }
  ws.off = mark;
}

void sfe(hipStream_t st, WS& ws, const SFEp& p, const float* xyz, int S,
         const float* feats, int Cf, const float* cost, int Cc,
         const float* flow, float* feat_out, float* flow_out) {
  size_t mark = ws.off;
  int C0 = Cf + Cc + (flow ? 3 : 0);
  float* x = ws.f((size_t)S * C0);
  if (flow)
    concat3_kernel<<<cdiv(S * C0, 128), 128, 0, st>>>(feats, Cf, cost, Cc, flow, 3, S, x);
  else
    concat2_kernel<<<cdiv(S * C0, 128), 128, 0, st>>>(feats, Cf, cost, Cc, S, x);
  int* idx9 = ws.i((size_t)S * 9);
  knn(st, 9, xyz, S, xyz, S, idx9);
  float* h1 = ws.f((size_t)S * 128);
  pconv_core(st, ws, p.pc[0], xyz, x, C0, idx9, xyz, S, 9, 128, h1);
  float* h2 = ws.f((size_t)S * 128);
  pconv_core(st, ws, p.pc[1], xyz, h1, 128, idx9, xyz, S, 9, 128, h2);
  float* h3 = ws.f((size_t)S * 128);
  linear(st, h2, S, 128, p.mlp[0], 128, h3, 1);
  linear(st, h3, S, 128, p.mlp[1], 64, feat_out, 1);
  linear(st, feat_out, S, 64, p.fc, 3, flow_out, 3);
  ws.off = mark;
}

}  // namespace

extern "C" void kernel_launch(void* const* d_in, const int* in_sizes, int n_in,
                              void* d_out, int out_size, void* d_ws,
                              size_t ws_size, hipStream_t stream) {
  if (n_in < 214 || !d_ws) return;
  const float* xyz1 = (const float*)d_in[0];
  const float* xyz2 = (const float*)d_in[1];
  const float* color1 = (const float*)d_in[2];
  const float* color2 = (const float*)d_in[3];

  // Decode parameter leaves (JAX pytree order: dicts by sorted key).
  int cur = 4;
  auto nxt = [&]() { return (const float*)d_in[cur++]; };
  auto rdLin = [&]() { Lin L; L.b = nxt(); L.w = nxt(); return L; };
  auto rdWN = [&]() { WN w; for (int i = 0; i < 3; ++i) w.l[i] = rdLin(); return w; };
  auto rdPC = [&]() { PConv p; p.lin_b = nxt(); p.lin_w = nxt(); p.wn = rdWN(); return p; };
  auto rdPCF = [&]() { PCF p; p.mlp[0] = rdLin(); p.mlp[1] = rdLin(); p.wn1 = rdWN(); p.wn2 = rdWN(); return p; };
  auto rdSFE = [&]() { SFEp s; s.fc = rdLin(); s.mlp[0] = rdLin(); s.mlp[1] = rdLin(); s.pc[0] = rdPC(); s.pc[1] = rdPC(); return s; };

  PCF p_cost0 = rdPCF(), p_cost1 = rdPCF(), p_cost2 = rdPCF(), p_cost3 = rdPCF();
  Lin deconv1_0 = rdLin(), deconv2_1 = rdLin(), deconv3_2 = rdLin(), deconv4_3 = rdLin();
  SFEp p_flow0 = rdSFE(), p_flow1 = rdSFE(), p_flow2 = rdSFE(), p_flow3 = rdSFE();
  Lin level0 = rdLin(), level0_1 = rdLin(), level0_2 = rdLin();
  PConv level1 = rdPC(); Lin level1_0 = rdLin(), level1_1 = rdLin();
  PConv level2 = rdPC(); Lin level2_0 = rdLin(), level2_1 = rdLin();
  PConv level3 = rdPC(); Lin level3_0 = rdLin(), level3_1 = rdLin();
  PConv level4 = rdPC();

  WS ws{(char*)d_ws, ws_size, 0};
  hipStream_t st = stream;
  const int N = 8192, S1 = 2048, S2 = 512, S3 = 256, S4 = 64;

  // ---- level 0 -------------------------------------------------------------
  float* t0 = ws.f((size_t)N * 32);
  float* f1_0 = ws.f((size_t)N * 32);
  float* f2_0 = ws.f((size_t)N * 32);
  float* f1_0_1 = ws.f((size_t)N * 64);
  float* f2_0_1 = ws.f((size_t)N * 64);
  linear(st, color1, N, 3, level0, 32, t0, 1);
  linear(st, t0, N, 32, level0_1, 32, f1_0, 1);
  linear(st, color2, N, 3, level0, 32, t0, 1);
  linear(st, t0, N, 32, level0_1, 32, f2_0, 1);
  linear(st, f1_0, N, 32, level0_2, 64, f1_0_1, 1);
  linear(st, f2_0, N, 32, level0_2, 64, f2_0_1, 1);

  // ---- level 1 -------------------------------------------------------------
  float* pc1_1 = ws.f((size_t)S1 * 3); float* f1_1 = ws.f((size_t)S1 * 64);
  float* pc2_1 = ws.f((size_t)S1 * 3); float* f2_1 = ws.f((size_t)S1 * 64);
  pointconv_d(st, ws, level1, S1, 16, xyz1, N, f1_0_1, 64, 64, pc1_1, f1_1);
  pointconv_d(st, ws, level1, S1, 16, xyz2, N, f2_0_1, 64, 64, pc2_1, f2_1);
  float* f1_1_2 = ws.f((size_t)S1 * 128); float* f2_1_2 = ws.f((size_t)S1 * 128);
  float* t1 = ws.f((size_t)S1 * 64);
  linear(st, f1_1, S1, 64, level1_0, 64, t1, 1);
  linear(st, t1, S1, 64, level1_1, 128, f1_1_2, 1);
  linear(st, f2_1, S1, 64, level1_0, 64, t1, 1);
  linear(st, t1, S1, 64, level1_1, 128, f2_1_2, 1);

  // ---- level 2 -------------------------------------------------------------
  float* pc1_2 = ws.f((size_t)S2 * 3); float* f1_2 = ws.f((size_t)S2 * 128);
  float* pc2_2 = ws.f((size_t)S2 * 3); float* f2_2 = ws.f((size_t)S2 * 128);
  pointconv_d(st, ws, level2, S2, 16, pc1_1, S1, f1_1_2, 128, 128, pc1_2, f1_2);
  pointconv_d(st, ws, level2, S2, 16, pc2_1, S1, f2_1_2, 128, 128, pc2_2, f2_2);
  float* f1_2_3 = ws.f((size_t)S2 * 256); float* f2_2_3 = ws.f((size_t)S2 * 256);
  float* t2 = ws.f((size_t)S2 * 128);
  linear(st, f1_2, S2, 128, level2_0, 128, t2, 1);
  linear(st, t2, S2, 128, level2_1, 256, f1_2_3, 1);
  linear(st, f2_2, S2, 128, level2_0, 128, t2, 1);
  linear(st, t2, S2, 128, level2_1, 256, f2_2_3, 1);

  // ---- level 3 -------------------------------------------------------------
  float* pc1_3 = ws.f((size_t)S3 * 3); float* f1_3 = ws.f((size_t)S3 * 256);
  float* pc2_3 = ws.f((size_t)S3 * 3); float* f2_3 = ws.f((size_t)S3 * 256);
  pointconv_d(st, ws, level3, S3, 16, pc1_2, S2, f1_2_3, 256, 256, pc1_3, f1_3);
  pointconv_d(st, ws, level3, S3, 16, pc2_2, S2, f2_2_3, 256, 256, pc2_3, f2_3);
  float* f1_3_4 = ws.f((size_t)S3 * 512); float* f2_3_4 = ws.f((size_t)S3 * 512);
  float* t3 = ws.f((size_t)S3 * 256);
  linear(st, f1_3, S3, 256, level3_0, 256, t3, 1);
  linear(st, t3, S3, 256, level3_1, 512, f1_3_4, 1);
  linear(st, f2_3, S3, 256, level3_0, 256, t3, 1);
  linear(st, t3, S3, 256, level3_1, 512, f2_3_4, 1);

  // ---- level 4 -------------------------------------------------------------
  float* pc1_4 = ws.f((size_t)S4 * 3); float* f1_4 = ws.f((size_t)S4 * 256);
  float* pc2_4 = ws.f((size_t)S4 * 3); float* f2_4 = ws.f((size_t)S4 * 256);
  pointconv_d(st, ws, level4, S4, 16, pc1_3, S3, f1_3_4, 512, 256, pc1_4, f1_4);
  pointconv_d(st, ws, level4, S4, 16, pc2_3, S3, f2_3_4, 512, 256, pc2_4, f2_4);

  // ---- deconv 4->3, concat -------------------------------------------------
  float* u43 = ws.f((size_t)S3 * 256);
  float* f1_4_3 = ws.f((size_t)S3 * 64); float* f2_4_3 = ws.f((size_t)S3 * 64);
  upsample(st, ws, pc1_3, S3, pc1_4, S4, f1_4, 256, u43);
  linear(st, u43, S3, 256, deconv4_3, 64, f1_4_3, 1);
  upsample(st, ws, pc2_3, S3, pc2_4, S4, f2_4, 256, u43);
  linear(st, u43, S3, 256, deconv4_3, 64, f2_4_3, 1);
  float* c_f1_3 = ws.f((size_t)S3 * 320); float* c_f2_3 = ws.f((size_t)S3 * 320);
  concat2_kernel<<<cdiv(S3 * 320, 128), 128, 0, st>>>(f1_3, 256, f1_4_3, 64, S3, c_f1_3);
  concat2_kernel<<<cdiv(S3 * 320, 128), 128, 0, st>>>(f2_3, 256, f2_4_3, 64, S3, c_f2_3);

  // ---- cost3 + flow3 -------------------------------------------------------
  float* cost3 = ws.f((size_t)S3 * 256);
  pcflow(st, ws, p_cost3, 32, pc1_3, S3, pc2_3, S3, c_f1_3, c_f2_3, 320, 256, 256, cost3);
  float* feat3 = ws.f((size_t)S3 * 64); float* flow3 = ws.f((size_t)S3 * 3);
  sfe(st, ws, p_flow3, pc1_3, S3, f1_3, 256, cost3, 256, nullptr, feat3, flow3);

  // ---- deconv 3->2, 2->1, 1->0 + concats -----------------------------------
  float* u32 = ws.f((size_t)S2 * 256);
  float* f1_3_2 = ws.f((size_t)S2 * 64); float* f2_3_2 = ws.f((size_t)S2 * 64);
  upsample(st, ws, pc1_2, S2, pc1_3, S3, f1_3, 256, u32);
  linear(st, u32, S2, 256, deconv3_2, 64, f1_3_2, 1);
  upsample(st, ws, pc2_2, S2, pc2_3, S3, f2_3, 256, u32);
  linear(st, u32, S2, 256, deconv3_2, 64, f2_3_2, 1);
  float* c_f1_2 = ws.f((size_t)S2 * 192); float* c_f2_2 = ws.f((size_t)S2 * 192);
  concat2_kernel<<<cdiv(S2 * 192, 128), 128, 0, st>>>(f1_2, 128, f1_3_2, 64, S2, c_f1_2);
  concat2_kernel<<<cdiv(S2 * 192, 128), 128, 0, st>>>(f2_2, 128, f2_3_2, 64, S2, c_f2_2);

  float* u21 = ws.f((size_t)S1 * 128);
  float* f1_2_1 = ws.f((size_t)S1 * 32); float* f2_2_1 = ws.f((size_t)S1 * 32);
  upsample(st, ws, pc1_1, S1, pc1_2, S2, f1_2, 128, u21);
  linear(st, u21, S1, 128, deconv2_1, 32, f1_2_1, 1);
  upsample(st, ws, pc2_1, S1, pc2_2, S2, f2_2, 128, u21);
  linear(st, u21, S1, 128, deconv2_1, 32, f2_2_1, 1);
  float* c_f1_1 = ws.f((size_t)S1 * 96); float* c_f2_1 = ws.f((size_t)S1 * 96);
  concat2_kernel<<<cdiv(S1 * 96, 128), 128, 0, st>>>(f1_1, 64, f1_2_1, 32, S1, c_f1_1);
  concat2_kernel<<<cdiv(S1 * 96, 128), 128, 0, st>>>(f2_1, 64, f2_2_1, 32, S1, c_f2_1);

  float* u10 = ws.f((size_t)N * 64);
  float* f1_1_0 = ws.f((size_t)N * 32); float* f2_1_0 = ws.f((size_t)N * 32);
  upsample(st, ws, xyz1, N, pc1_1, S1, f1_1, 64, u10);
  linear(st, u10, N, 64, deconv1_0, 32, f1_1_0, 1);
  upsample(st, ws, xyz2, N, pc2_1, S1, f2_1, 64, u10);
  linear(st, u10, N, 64, deconv1_0, 32, f2_1_0, 1);
  float* c_f1_0 = ws.f((size_t)N * 64); float* c_f2_0 = ws.f((size_t)N * 64);
  concat2_kernel<<<cdiv(N * 64, 128), 128, 0, st>>>(f1_0, 32, f1_1_0, 32, N, c_f1_0);
  concat2_kernel<<<cdiv(N * 64, 128), 128, 0, st>>>(f2_0, 32, f2_1_0, 32, N, c_f2_0);

  // ---- level-2 flow --------------------------------------------------------
  float* up_flow2 = ws.f((size_t)S2 * 3);
  upsample(st, ws, pc1_2, S2, pc1_3, S3, flow3, 3, up_flow2);
  float* pc2_2w = ws.f((size_t)S2 * 3);
  warp(st, ws, pc1_2, S2, pc2_2, S2, up_flow2, pc2_2w);
  float* cost2 = ws.f((size_t)S2 * 128);
  pcflow(st, ws, p_cost2, 32, pc1_2, S2, pc2_2w, S2, c_f1_2, c_f2_2, 192, 128, 128, cost2);
  float* feat3_up = ws.f((size_t)S2 * 64);
  upsample(st, ws, pc1_2, S2, pc1_3, S3, feat3, 64, feat3_up);
  float* new_f1_2 = ws.f((size_t)S2 * 192);
  concat2_kernel<<<cdiv(S2 * 192, 128), 128, 0, st>>>(f1_2, 128, feat3_up, 64, S2, new_f1_2);
  float* feat2 = ws.f((size_t)S2 * 64); float* rf2 = ws.f((size_t)S2 * 3);
  sfe(st, ws, p_flow2, pc1_2, S2, new_f1_2, 192, cost2, 128, up_flow2, feat2, rf2);
  float* flow2 = ws.f((size_t)S2 * 3);
  add_kernel<<<cdiv(S2 * 3, 128), 128, 0, st>>>(up_flow2, rf2, S2 * 3, flow2);

  // ---- level-1 flow --------------------------------------------------------
  float* up_flow1 = ws.f((size_t)S1 * 3);
  upsample(st, ws, pc1_1, S1, pc1_2, S2, flow2, 3, up_flow1);
  float* pc2_1w = ws.f((size_t)S1 * 3);
  warp(st, ws, pc1_1, S1, pc2_1, S1, up_flow1, pc2_1w);
  float* cost1 = ws.f((size_t)S1 * 64);
  pcflow(st, ws, p_cost1, 32, pc1_1, S1, pc2_1w, S1, c_f1_1, c_f2_1, 96, 64, 64, cost1);
  float* feat2_up = ws.f((size_t)S1 * 64);
  upsample(st, ws, pc1_1, S1, pc1_2, S2, feat2, 64, feat2_up);
  float* new_f1_1 = ws.f((size_t)S1 * 128);
  concat2_kernel<<<cdiv(S1 * 128, 128), 128, 0, st>>>(f1_1, 64, feat2_up, 64, S1, new_f1_1);
  float* feat1 = ws.f((size_t)S1 * 64); float* rf1 = ws.f((size_t)S1 * 3);
  sfe(st, ws, p_flow1, pc1_1, S1, new_f1_1, 128, cost1, 64, up_flow1, feat1, rf1);
  float* flow1 = ws.f((size_t)S1 * 3);
  add_kernel<<<cdiv(S1 * 3, 128), 128, 0, st>>>(up_flow1, rf1, S1 * 3, flow1);

  // ---- level-0 flow --------------------------------------------------------
  float* up_flow0 = ws.f((size_t)N * 3);
  upsample(st, ws, xyz1, N, pc1_1, S1, flow1, 3, up_flow0);
  float* pc2_0w = ws.f((size_t)N * 3);
  warp(st, ws, xyz1, N, xyz2, N, up_flow0, pc2_0w);
  float* cost0 = ws.f((size_t)N * 32);
  pcflow(st, ws, p_cost0, 32, xyz1, N, pc2_0w, N, c_f1_0, c_f2_0, 64, 32, 32, cost0);
  float* feat1_up = ws.f((size_t)N * 64);
  upsample(st, ws, xyz1, N, pc1_1, S1, feat1, 64, feat1_up);
  float* new_f1_0 = ws.f((size_t)N * 96);
  concat2_kernel<<<cdiv(N * 96, 128), 128, 0, st>>>(f1_0, 32, feat1_up, 64, N, new_f1_0);
  float* feat0 = ws.f((size_t)N * 64); float* rf0 = ws.f((size_t)N * 3);
  sfe(st, ws, p_flow0, xyz1, N, new_f1_0, 96, cost0, 32, up_flow0, feat0, rf0);

  // flow0 = up_flow0 + rf0 -> output
  add_kernel<<<cdiv(N * 3, 128), 128, 0, st>>>(up_flow0, rf0, N * 3, (float*)d_out);
  (void)in_sizes; (void)out_size;
}
